// FullRNN_36910948942509
// MI455X (gfx1250) — matrix operand order repaired and checked by reading, not compiled, not run on hardware
//
#include <hip/hip_runtime.h>
#include <hip/hip_bf16.h>

typedef __attribute__((ext_vector_type(16))) __bf16 v16bf;
typedef __attribute__((ext_vector_type(8)))  __bf16 v8bf;
typedef __attribute__((ext_vector_type(8)))  float  v8f;
typedef __attribute__((ext_vector_type(4)))  unsigned int v4u;
typedef __attribute__((ext_vector_type(8)))  int    v8i;
typedef __attribute__((ext_vector_type(4)))  int    v4i;

#define B_  64
#define T_  2048
#define I_  128
#define H_  512
#define O_  64
#define ALPHA_    0.1f
#define TAUSCALE_ 10.0f

// Workspace layout:
//   [0)        weight B-fragments (bf16): win 65536 | wrec 262144 | ro 32768
//   [720896)   params f32[3*512]
//   [1MB)      xfrag bf16 [g=4][t=2048][2048]  (A-fragment order, 32 MB)
#define WIN_OFF 0
#define WR_OFF  65536
#define RO_OFF  (65536 + 262144)
#define FRAG_BYTES 720896
#define XFRAG_BYTE_OFF (1u << 20)

// ---------------------------------------------------------------------------
// TDM: 1-D tile of `ndwords` dwords, global -> LDS.  D# per CDNA5 ISA ch.8.
// Group0: count=1 | lds_addr | global_addr | type=2.  Group1: data_size=4B,
// tensor_dim0 = tile_dim0 = ndwords, tensor_dim1 = tile_dim1 = 1.
// 6-arg builtin (clang-23 lane): (v4u g0, v8i g1, v4i g2, v4i g3, v8i g4, cpol)
// ---------------------------------------------------------------------------
__device__ __forceinline__ void tdm_load_1d(unsigned lds_off,
                                            unsigned long long gaddr,
                                            unsigned ndwords)
{
    v4u g0;
    g0[0] = 1u;                                                   // count=1, user
    g0[1] = lds_off;                                              // LDS byte addr
    g0[2] = (unsigned)(gaddr & 0xFFFFFFFFu);                      // addr[31:0]
    g0[3] = (unsigned)((gaddr >> 32) & 0x1FFFFFFu) | (2u << 30);  // addr[56:32]|type=2

    v8i g1;
    g1[0] = (int)(2u << 16);                      // data_size=4B, no mcast/pad/iter
    g1[1] = (int)((ndwords & 0xFFFFu) << 16);     // tensor_dim0[15:0]
    g1[2] = (int)((ndwords >> 16) | (1u << 16)); // tensor_dim0[31:16] | tensor_dim1=1
    g1[3] = (int)(ndwords << 16);                 // tile_dim0 (16-bit)
    g1[4] = 1;                                    // tile_dim1=1, tile_dim2=0
    g1[5] = (int)ndwords;                         // tensor_dim0_stride
    g1[6] = 0;
    g1[7] = 0;

    v4i z4 = {0, 0, 0, 0};
    v8i z8 = {0, 0, 0, 0, 0, 0, 0, 0};
    __builtin_amdgcn_tensor_load_to_lds(g0, g1, z4, z4, z8, 0);
}

// ---------------------------------------------------------------------------
// Prep 1: weights -> bf16 WMMA B-fragment order + per-column params.
// ---------------------------------------------------------------------------
__global__ void FullRNN_prep_kernel(const float* __restrict__ Win,
                                    const float* __restrict__ Wrec,
                                    const float* __restrict__ Ro,
                                    const float* __restrict__ g_,
                                    const float* __restrict__ tau_,
                                    const float* __restrict__ th_,
                                    __bf16* __restrict__ frags,
                                    float* __restrict__ params)
{
    int idx    = blockIdx.x * blockDim.x + threadIdx.x;
    int stride = gridDim.x * blockDim.x;
    const int NW_IN = 4  * 32 * 32 * 16;
    const int NW_RC = 16 * 32 * 32 * 16;
    const int NW_RO = 16 * 4  * 32 * 16;

    for (int f = idx; f < NW_IN + NW_RC + NW_RO; f += stride) {
        if (f < NW_IN) {
            int j = f & 15, lane = (f >> 4) & 31, nt = (f >> 9) & 31, kt = f >> 14;
            int n = nt * 16 + (lane & 15);
            int k = kt * 32 + (lane >> 4) * 8 + (j & 7) + ((j >= 8) ? 16 : 0);
            frags[f] = (__bf16)Win[n * I_ + k];
        } else if (f < NW_IN + NW_RC) {
            int g2 = f - NW_IN;
            int j = g2 & 15, lane = (g2 >> 4) & 31, nt = (g2 >> 9) & 31, kt = g2 >> 14;
            int n = nt * 16 + (lane & 15);
            int k = kt * 32 + (lane >> 4) * 8 + (j & 7) + ((j >= 8) ? 16 : 0);
            frags[f] = (__bf16)Wrec[n * H_ + k];
        } else {
            int g2 = f - NW_IN - NW_RC;
            int j = g2 & 15, lane = (g2 >> 4) & 31, nt = (g2 >> 9) & 3, kt = g2 >> 11;
            int n = nt * 16 + (lane & 15);
            int k = kt * 32 + (lane >> 4) * 8 + (j & 7) + ((j >= 8) ? 16 : 0);
            frags[f] = (__bf16)Ro[n * H_ + k];
        }
    }
    for (int h = idx; h < H_; h += stride) {
        params[h]          = 1.f / (1.f + expf(-g_[h]));
        params[H_ + h]     = ALPHA_ * TAUSCALE_ * (1.f / (1.f + expf(-tau_[h])));
        params[2 * H_ + h] = th_[h];
    }
}

// ---------------------------------------------------------------------------
// Prep 2: x (f32) -> bf16 A-fragment order:
//   xfrag[g][t][kt(4)][lane(32)][j(16)], k = kt*32 + (lane/16)*8 + (j&7) + (j>=8)*16
// ---------------------------------------------------------------------------
__global__ void FullRNN_xfrag_kernel(const float* __restrict__ x,
                                     __bf16* __restrict__ xfrag)
{
    int f = blockIdx.x * blockDim.x + threadIdx.x;   // 4*2048*2048 = 16,777,216
    int j    = f & 15;
    int lane = (f >> 4) & 31;
    int kt   = (f >> 9) & 3;
    int t    = (f >> 11) & 2047;
    int g    = f >> 22;
    int b    = g * 16 + (lane & 15);
    int k    = kt * 32 + ((lane >> 4) << 3) + (j & 7) + ((j >= 8) ? 16 : 0);
    xfrag[f] = (__bf16)x[((size_t)b * T_ + t) * I_ + k];
}

// ---------------------------------------------------------------------------
// Persistent scan: 4 WGs x 16 batch rows, 8 wave32s. Two s_barriers per step.
// TDM double-buffers the next step's x fragment tile into LDS.
// ---------------------------------------------------------------------------
__global__ __launch_bounds__(256, 1)
void FullRNN_scan_kernel(const __bf16* __restrict__ xfrag,
                         const __bf16* __restrict__ frags,
                         const float* __restrict__ params,
                         const float* __restrict__ rb,
                         const float* __restrict__ rscale_p,
                         float* __restrict__ out)
{
    __shared__ float  hF[16][H_];                       // 32 KB f32 state
    __shared__ __bf16 hB[2][16][H_];                    // 32 KB bf16 operand (dbl buf)
    __shared__ __align__(128) __bf16 xbuf[2][2048];     //  8 KB x frag tiles (dbl buf)
    __shared__ float  pGain[H_], pSS[H_], pThr[H_];

    const int tid  = threadIdx.x;
    const int lane = tid & 31;
    const int wave = tid >> 5;
    const int b0   = blockIdx.x * 16;
    const float rscale = rscale_p[0];

    const __bf16* winF = frags + WIN_OFF;
    const __bf16* wrF  = frags + WR_OFF;
    const __bf16* roF  = frags + RO_OFF;
    const __bf16* xg   = xfrag + (size_t)blockIdx.x * T_ * 2048;

    for (int i = tid; i < H_; i += 256) {
        pGain[i] = params[i];
        pSS[i]   = params[H_ + i];
        pThr[i]  = params[2 * H_ + i];
    }
    for (int i = tid; i < 16 * H_; i += 256) {
        ((float*)hF)[i]        = 0.f;
        ((__bf16*)hB[0])[i]    = (__bf16)0.f;
    }
    if (wave == 7) {   // preload x tile for t=0 via TDM, wait on TENSORcnt
        tdm_load_1d((unsigned)(uintptr_t)&xbuf[0][0],
                    (unsigned long long)(uintptr_t)xg, 1024);
        __builtin_amdgcn_s_wait_tensorcnt(0);
    }
    __syncthreads();

    for (int t = 0; t < T_; ++t) {
        const int cur = t & 1, nxt = cur ^ 1;

        // ---- async stage next x tile (wave 7 only; EXEC-uniform branch) ----
        if (wave == 7) {
            int tn = (t + 1 < T_) ? t + 1 : t;
            tdm_load_1d((unsigned)(uintptr_t)&xbuf[nxt][0],
                        (unsigned long long)(uintptr_t)(xg + (size_t)tn * 2048), 1024);
        }

        // ---- acc = x_t @ Win^T + h @ Wrec^T  (each wave: 4 N-tiles) ----
        v8f acc[4] = {};

#pragma unroll
        for (int kt = 0; kt < 4; ++kt) {               // input projection K=128
            int base = (kt * 32 + lane) * 16;
            v8bf lo = *(const v8bf*)&xbuf[cur][base];
            v8bf hi = *(const v8bf*)&xbuf[cur][base + 8];
            v16bf a;
#pragma unroll
            for (int j = 0; j < 8; ++j) { a[j] = lo[j]; a[8 + j] = hi[j]; }
#pragma unroll
            for (int nt = 0; nt < 4; ++nt) {
                int ntile = wave * 4 + nt;
                v16bf bfr = *(const v16bf*)(winF + (((kt * 32 + ntile) * 32) + lane) * 16);
                acc[nt] = __builtin_amdgcn_wmma_f32_16x16x32_bf16(
                    false, a, false, bfr, (short)0, acc[nt], false, false);
            }
        }
#pragma unroll 4
        for (int kt = 0; kt < 16; ++kt) {              // recurrent K=512
            int m  = lane & 15;
            int kb = kt * 32 + (lane >> 4) * 8;
            v8bf lo = *(const v8bf*)&hB[cur][m][kb];
            v8bf hi = *(const v8bf*)&hB[cur][m][kb + 16];
            v16bf a;
#pragma unroll
            for (int j = 0; j < 8; ++j) { a[j] = lo[j]; a[8 + j] = hi[j]; }
#pragma unroll
            for (int nt = 0; nt < 4; ++nt) {
                int ntile = wave * 4 + nt;
                v16bf bfr = *(const v16bf*)(wrF + (((kt * 32 + ntile) * 32) + lane) * 16);
                acc[nt] = __builtin_amdgcn_wmma_f32_16x16x32_bf16(
                    false, a, false, bfr, (short)0, acc[nt], false, false);
            }
        }

        // ---- elementwise: h += (gain*relu(acc - thr) - h)*ss -> hB[nxt] ----
#pragma unroll
        for (int nt = 0; nt < 4; ++nt) {
            int col  = (wave * 4 + nt) * 16 + (lane & 15);
            float gn = pGain[col], ss = pSS[col], th = pThr[col];
#pragma unroll
            for (int r = 0; r < 8; ++r) {
                int m = r + ((lane >> 4) << 3);
                float pre  = acc[nt][r] - th;
                float act  = gn * (pre > 0.f ? pre : 0.f);
                float hold = hF[m][col];
                float hnew = hold + (act - hold) * ss;
                hF[m][col]      = hnew;
                hB[nxt][m][col] = (__bf16)hnew;
            }
        }

        if (wave == 7) __builtin_amdgcn_s_wait_tensorcnt(0);  // xbuf[nxt] landed
        __syncthreads();   // first barrier/step: publishes hB[nxt] + xbuf[nxt]

        // ---- readout (waves 0..3); waves 4..7 prefetch xfrag stream ----
        if (wave < 4) {
            v8f ro = {};
#pragma unroll 4
            for (int kt = 0; kt < 16; ++kt) {
                int m  = lane & 15;
                int kb = kt * 32 + (lane >> 4) * 8;
                v8bf lo = *(const v8bf*)&hB[nxt][m][kb];
                v8bf hi = *(const v8bf*)&hB[nxt][m][kb + 16];
                v16bf a;
#pragma unroll
                for (int j = 0; j < 8; ++j) { a[j] = lo[j]; a[8 + j] = hi[j]; }
                v16bf bfr = *(const v16bf*)(roF + (((kt * 4 + wave) * 32) + lane) * 16);
                ro = __builtin_amdgcn_wmma_f32_16x16x32_bf16(
                    false, a, false, bfr, (short)0, ro, false, false);
            }
            int o    = wave * 16 + (lane & 15);
            float bv = rb[o];
#pragma unroll
            for (int r = 0; r < 8; ++r) {
                int m = r + ((lane >> 4) << 3);
                out[((size_t)(b0 + m) * T_ + t) * O_ + o] = rscale * (ro[r] + bv);
            }
        } else {
            int tp  = (t + 4 < T_) ? t + 4 : T_ - 1;   // warm L2 a few steps ahead
            int l64 = tid - 128;                       // 0..127 -> 2 lines each
            __builtin_prefetch(xg + (size_t)tp * 2048 + l64 * 32, 0, 1);
        }
        __syncthreads();
    }
}

extern "C" void kernel_launch(void* const* d_in, const int* in_sizes, int n_in,
                              void* d_out, int out_size, void* d_ws, size_t ws_size,
                              hipStream_t stream) {
    (void)in_sizes; (void)n_in; (void)out_size; (void)ws_size;
    const float* x    = (const float*)d_in[0];
    const float* Win  = (const float*)d_in[1];
    const float* g    = (const float*)d_in[2];
    const float* tau  = (const float*)d_in[3];
    const float* th   = (const float*)d_in[4];
    const float* Wrec = (const float*)d_in[5];
    const float* Ro   = (const float*)d_in[6];
    const float* rb   = (const float*)d_in[7];
    const float* rsc  = (const float*)d_in[8];

    __bf16* frags  = (__bf16*)d_ws;
    float*  params = (float*)((char*)d_ws + FRAG_BYTES);
    __bf16* xfrag  = (__bf16*)((char*)d_ws + XFRAG_BYTE_OFF);

    FullRNN_prep_kernel<<<512, 256, 0, stream>>>(Win, Wrec, Ro, g, tau, th, frags, params);
    FullRNN_xfrag_kernel<<<(4 * T_ * 2048) / 256, 256, 0, stream>>>(x, xfrag);
    FullRNN_scan_kernel<<<4, 256, 0, stream>>>(xfrag, frags, params, rb, rsc, (float*)d_out);
}
